// MultiHeadAttention_60748017435079
// MI455X (gfx1250) — matrix operand order repaired and checked by reading, not compiled, not run on hardware
//
#include <hip/hip_runtime.h>
#include <hip/hip_bf16.h>

#define B_  2
#define S_  2048
#define D_  1024
#define H_  16
#define DK_ 64

typedef __attribute__((ext_vector_type(16))) __bf16 v16bf;
typedef __attribute__((ext_vector_type(8)))  __bf16 v8bf;
typedef __attribute__((ext_vector_type(8)))  float  v8f;

// ---------------------------------------------------------------------------
// WMMA helpers. Layouts per CDNA5 ISA 7.12.2 (wave32):
//  A (16x32 bf16): lane l -> row l&15; half=l>>4: K chunks [half*8,+8) and
//                  [16+half*8,+8)  -> two contiguous 16B loads per lane.
//  B (32x16 bf16): lane l -> column l&15; K chunks [half*8,+8) and
//                  [16+half*8,+8) -> two contiguous 16B loads per lane (N-major).
//  C/D (16x16 f32): element (m,n) in lane n+16*(m>=8), vgpr m%8.
// ---------------------------------------------------------------------------
static __device__ __forceinline__ v16bf load_a_tile(const __bf16* p, int ld,
                                                    int row0, int k0, int lane) {
    const int r    = row0 + (lane & 15);
    const int half = lane >> 4;
    const __bf16* base = p + (long)r * ld + k0 + half * 8;
    v8bf lo = *(const v8bf*)(base);
    v8bf hi = *(const v8bf*)(base + 16);
    v16bf out;
#pragma unroll
    for (int i = 0; i < 8; ++i) { out[i] = lo[i]; out[i + 8] = hi[i]; }
    return out;
}

// p is N-major: p[n*ld + k]; returns B tile with columns n0..n0+15, K k0..k0+31
static __device__ __forceinline__ v16bf load_b_tile(const __bf16* p, int ld,
                                                    int n0, int k0, int lane) {
    const __bf16* base = p + (long)(n0 + (lane & 15)) * ld + k0 + (lane >> 4) * 8;
    v8bf lo = *(const v8bf*)(base);
    v8bf hi = *(const v8bf*)(base + 16);
    v16bf out;
#pragma unroll
    for (int i = 0; i < 8; ++i) { out[i] = lo[i]; out[i + 8] = hi[i]; }
    return out;
}

static __device__ __forceinline__ v8f wmma_bf16(v16bf a, v16bf b, v8f c) {
    // v_wmma_f32_16x16x32_bf16
    return __builtin_amdgcn_wmma_f32_16x16x32_bf16(false, a, false, b,
                                                   (short)0, c, false, false);
}

// ---------------------------------------------------------------------------
// Precision conversion kernels
// ---------------------------------------------------------------------------
__global__ void __launch_bounds__(256)
cvt_bf16_kernel(const float* __restrict__ src, __bf16* __restrict__ dst, int n) {
    int i = blockIdx.x * blockDim.x + threadIdx.x;
    if (i < n) dst[i] = (__bf16)src[i];
}

// Wt[n*D + k] = (bf16) W[k*D + n]   (N-major for B-tile loads)
__global__ void __launch_bounds__(256)
transpose_w_kernel(const float* __restrict__ W, __bf16* __restrict__ Wt) {
    int i = blockIdx.x * blockDim.x + threadIdx.x;
    int n = i / D_, k = i % D_;
    Wt[i] = (__bf16)W[(long)k * D_ + n];
}

// ---------------------------------------------------------------------------
// Fused QKV projection. One wave computes a 64x64 tile of one of Q/K/V:
// 4 B tiles kept resident, 4 A tiles streamed -> 16 b128 loads per 16 WMMAs.
// __launch_bounds__(256) => 8 waves/WGP known at compile time, so the wave
// may use up to ~512 VGPRs: the 128-VGPR accumulator set stays in registers.
// Tasks: mat(3) x rowTiles64(64) x colTiles64(16) = 3072 waves.
// Q,K stored per-head row-major [BH][S][64]; V stored transposed [BH][64][S].
// ---------------------------------------------------------------------------
__global__ void __launch_bounds__(256)
qkv_gemm_kernel(const __bf16* __restrict__ xb,
                const __bf16* __restrict__ Wtq,
                const __bf16* __restrict__ Wtk,
                const __bf16* __restrict__ Wtv,
                const float* __restrict__ bq,
                const float* __restrict__ bk,
                const float* __restrict__ bv,
                __bf16* __restrict__ Qb,
                __bf16* __restrict__ Kb,
                __bf16* __restrict__ Vt) {
    const int lane = threadIdx.x & 31;
    const int wave = threadIdx.x >> 5;
    const int task = blockIdx.x * 8 + wave;
    const int MT = (B_ * S_) / 64;   // 64
    const int NT = D_ / 64;          // 16
    const int mat = task / (MT * NT);
    const int rem = task % (MT * NT);
    const int rt = rem / NT;         // 64-row tile
    const int ct = rem % NT;         // 64-col tile
    const __bf16* Wt   = (mat == 0) ? Wtq : (mat == 1) ? Wtk : Wtv;
    const float*  bias = (mat == 0) ? bq  : (mat == 1) ? bk  : bv;

    v8f acc[4][4] = {{{}, {}, {}, {}}, {{}, {}, {}, {}},
                     {{}, {}, {}, {}}, {{}, {}, {}, {}}};

    for (int k0 = 0; k0 < D_; k0 += 32) {
        // gfx1250 prefetch (global_prefetch_b8) ~4 K-steps ahead
        if (k0 + 128 < D_) {
            __builtin_prefetch(Wt + (long)(ct * 64 + (lane & 15)) * D_ + k0 + 128, 0, 1);
            __builtin_prefetch(xb + (long)(rt * 64 + (lane & 31)) * D_ + k0 + 128, 0, 1);
        }
        const v16bf bt0 = load_b_tile(Wt, D_, ct * 64 + 0,  k0, lane);
        const v16bf bt1 = load_b_tile(Wt, D_, ct * 64 + 16, k0, lane);
        const v16bf bt2 = load_b_tile(Wt, D_, ct * 64 + 32, k0, lane);
        const v16bf bt3 = load_b_tile(Wt, D_, ct * 64 + 48, k0, lane);
#pragma unroll
        for (int i = 0; i < 4; ++i) {
            const v16bf a = load_a_tile(xb, D_, rt * 64 + i * 16, k0, lane);
            acc[i][0] = wmma_bf16(a, bt0, acc[i][0]);
            acc[i][1] = wmma_bf16(a, bt1, acc[i][1]);
            acc[i][2] = wmma_bf16(a, bt2, acc[i][2]);
            acc[i][3] = wmma_bf16(a, bt3, acc[i][3]);
        }
    }

    const int half = lane >> 4;
    const int col  = lane & 15;
#pragma unroll
    for (int i = 0; i < 4; ++i) {
#pragma unroll
        for (int t = 0; t < 4; ++t) {
#pragma unroll
            for (int v = 0; v < 8; ++v) {
                const int r = rt * 64 + i * 16 + v + 8 * half;  // token in [0,B*S)
                const int c = ct * 64 + t * 16 + col;           // feature in [0,D)
                const float val = acc[i][t][v] + bias[c];
                const int b = r / S_, s = r % S_;
                const int h = c / DK_, dk = c % DK_;
                const __bf16 bvv = (__bf16)val;
                if (mat == 0)
                    Qb[(((long)(b * H_ + h)) * S_ + s) * DK_ + dk] = bvv;
                else if (mat == 1)
                    Kb[(((long)(b * H_ + h)) * S_ + s) * DK_ + dk] = bvv;
                else
                    Vt[(((long)(b * H_ + h)) * DK_ + dk) * S_ + s] = bvv;
            }
        }
    }
}

// ---------------------------------------------------------------------------
// Flash attention: one wave per (b,h, 16-query block); 32 keys per step.
// 4 score WMMAs + online softmax (base-2) + LDS transpose (C->A layout)
// + 4 ctx WMMAs.
// ---------------------------------------------------------------------------
__global__ void __launch_bounds__(256)
attn_kernel(const __bf16* __restrict__ Qb,
            const __bf16* __restrict__ Kb,
            const __bf16* __restrict__ Vt,
            __bf16* __restrict__ ctxb) {
    __shared__ __attribute__((aligned(32))) __bf16 lds[8][16 * 32];

    const int lane = threadIdx.x & 31;
    const int wave = threadIdx.x >> 5;
    const int w = blockIdx.x * 8 + wave;
    const int qblocks = S_ / 16;            // 128
    const int bh = w / qblocks;
    const int q0 = (w % qblocks) * 16;
    const int b = bh / H_;
    const int h = bh % H_;

    const __bf16* Qh = Qb + (long)bh * S_ * DK_;
    const __bf16* Kh = Kb + (long)bh * S_ * DK_;
    const __bf16* Vh = Vt + (long)bh * DK_ * S_;
    __bf16* myLds = &lds[wave][0];

    const v16bf aQ0 = load_a_tile(Qh, DK_, q0, 0, lane);
    const v16bf aQ1 = load_a_tile(Qh, DK_, q0, 32, lane);

    float m[8], l[8];
    v8f acc[4] = {{}, {}, {}, {}};
#pragma unroll
    for (int v = 0; v < 8; ++v) { m[v] = -1e30f; l[v] = 0.f; }

    // base-2 softmax: scores scaled by (1/sqrt(DK)) * log2(e)
    const float sm_scale = 0.125f * 1.44269504f;
    const int half = lane >> 4;
    const int col  = lane & 15;

    for (int kk0 = 0; kk0 < S_; kk0 += 32) {
        // prefetch K rows / V columns two steps ahead (global_prefetch_b8)
        if (kk0 + 64 < S_) {
            __builtin_prefetch(Kh + (long)(kk0 + 64 + lane) * DK_, 0, 1);
            __builtin_prefetch(Vh + (long)(lane * 2) * S_ + kk0 + 64, 0, 1);
        }
        // ---- scores: two 16x16 key tiles, K-dim 64 = 2x32 ----
        v8f s0 = {}, s1 = {};
        s0 = wmma_bf16(aQ0, load_b_tile(Kh, DK_, kk0,       0,  lane), s0);
        s0 = wmma_bf16(aQ1, load_b_tile(Kh, DK_, kk0,       32, lane), s0);
        s1 = wmma_bf16(aQ0, load_b_tile(Kh, DK_, kk0 + 16,  0,  lane), s1);
        s1 = wmma_bf16(aQ1, load_b_tile(Kh, DK_, kk0 + 16,  32, lane), s1);

        // ---- online softmax; row stats replicate across the 16-lane half ----
#pragma unroll
        for (int v = 0; v < 8; ++v) {
            const float a0 = s0[v] * sm_scale;
            const float a1 = s1[v] * sm_scale;
            float tmax = fmaxf(a0, a1);
            tmax = fmaxf(tmax, __shfl_xor(tmax, 1, 32));
            tmax = fmaxf(tmax, __shfl_xor(tmax, 2, 32));
            tmax = fmaxf(tmax, __shfl_xor(tmax, 4, 32));
            tmax = fmaxf(tmax, __shfl_xor(tmax, 8, 32));
            const float mnew = fmaxf(m[v], tmax);
            const float corr = exp2f(m[v] - mnew);      // v_exp_f32
            const float p0 = exp2f(a0 - mnew);
            const float p1 = exp2f(a1 - mnew);
            float rs = p0 + p1;
            rs += __shfl_xor(rs, 1, 32);
            rs += __shfl_xor(rs, 2, 32);
            rs += __shfl_xor(rs, 4, 32);
            rs += __shfl_xor(rs, 8, 32);
            l[v] = l[v] * corr + rs;
            m[v] = mnew;
            acc[0][v] *= corr; acc[1][v] *= corr;
            acc[2][v] *= corr; acc[3][v] *= corr;
            // stage P (C layout) into LDS, row-major 16x32 bf16
            const int row = v + 8 * half;
            myLds[row * 32 + col]      = (__bf16)p0;
            myLds[row * 32 + 16 + col] = (__bf16)p1;
        }
        asm volatile("s_wait_dscnt 0" ::: "memory");

        // ---- reload P in A layout (two contiguous 16B LDS loads) ----
        const __bf16* pb = myLds + (long)col * 32 + half * 8;
        const v8bf plo = *(const v8bf*)(pb);
        const v8bf phi = *(const v8bf*)(pb + 16);
        v16bf aP;
#pragma unroll
        for (int i = 0; i < 8; ++i) { aP[i] = plo[i]; aP[i + 8] = phi[i]; }

        // ---- ctx += P x V  (N = 64 dv split into 4 tiles, K = 32 keys) ----
#pragma unroll
        for (int t = 0; t < 4; ++t) {
            v16bf bV = load_b_tile(Vh, S_, t * 16, kk0, lane);
            acc[t] = wmma_bf16(aP, bV, acc[t]);
        }
    }

    // ---- normalize and store ctx as bf16 [B*S][D] row-major ----
#pragma unroll
    for (int t = 0; t < 4; ++t) {
#pragma unroll
        for (int v = 0; v < 8; ++v) {
            const float val = acc[t][v] / l[v];
            const int s = q0 + v + 8 * half;
            const int c = h * DK_ + t * 16 + col;
            ctxb[((long)(b * S_ + s)) * D_ + c] = (__bf16)val;
        }
    }
}

// ---------------------------------------------------------------------------
// Output projection: out = ctx @ Wo + bo (fp32 result), 64x64 tile per wave.
// ---------------------------------------------------------------------------
__global__ void __launch_bounds__(256)
oproj_kernel(const __bf16* __restrict__ ctxb,
             const __bf16* __restrict__ Wto,
             const float* __restrict__ bo,
             float* __restrict__ out) {
    const int lane = threadIdx.x & 31;
    const int wave = threadIdx.x >> 5;
    const int task = blockIdx.x * 8 + wave;
    const int NT = D_ / 64;                 // 16
    const int rt = task / NT;               // 0..63
    const int ct = task % NT;

    v8f acc[4][4] = {{{}, {}, {}, {}}, {{}, {}, {}, {}},
                     {{}, {}, {}, {}}, {{}, {}, {}, {}}};

    for (int k0 = 0; k0 < D_; k0 += 32) {
        if (k0 + 128 < D_) {
            __builtin_prefetch(Wto  + (long)(ct * 64 + (lane & 15)) * D_ + k0 + 128, 0, 1);
            __builtin_prefetch(ctxb + (long)(rt * 64 + (lane & 31)) * D_ + k0 + 128, 0, 1);
        }
        const v16bf bt0 = load_b_tile(Wto, D_, ct * 64 + 0,  k0, lane);
        const v16bf bt1 = load_b_tile(Wto, D_, ct * 64 + 16, k0, lane);
        const v16bf bt2 = load_b_tile(Wto, D_, ct * 64 + 32, k0, lane);
        const v16bf bt3 = load_b_tile(Wto, D_, ct * 64 + 48, k0, lane);
#pragma unroll
        for (int i = 0; i < 4; ++i) {
            const v16bf a = load_a_tile(ctxb, D_, rt * 64 + i * 16, k0, lane);
            acc[i][0] = wmma_bf16(a, bt0, acc[i][0]);
            acc[i][1] = wmma_bf16(a, bt1, acc[i][1]);
            acc[i][2] = wmma_bf16(a, bt2, acc[i][2]);
            acc[i][3] = wmma_bf16(a, bt3, acc[i][3]);
        }
    }

    const int half = lane >> 4;
    const int col  = lane & 15;
#pragma unroll
    for (int i = 0; i < 4; ++i) {
#pragma unroll
        for (int t = 0; t < 4; ++t) {
#pragma unroll
            for (int v = 0; v < 8; ++v) {
                const int r = rt * 64 + i * 16 + v + 8 * half;
                const int c = ct * 64 + t * 16 + col;
                out[(long)r * D_ + c] = acc[i][t][v] + bo[c];
            }
        }
    }
}

// ---------------------------------------------------------------------------
// Launch
// ---------------------------------------------------------------------------
extern "C" void kernel_launch(void* const* d_in, const int* in_sizes, int n_in,
                              void* d_out, int out_size, void* d_ws, size_t ws_size,
                              hipStream_t stream) {
    const float* x  = (const float*)d_in[0];
    const float* Wq = (const float*)d_in[1];
    const float* bq = (const float*)d_in[2];
    const float* Wk = (const float*)d_in[3];
    const float* bk = (const float*)d_in[4];
    const float* Wv = (const float*)d_in[5];
    const float* bv = (const float*)d_in[6];
    const float* Wo = (const float*)d_in[7];
    const float* bo = (const float*)d_in[8];
    float* out = (float*)d_out;

    char* ws = (char*)d_ws;
    const size_t MB = 1024 * 1024;
    __bf16* xb   = (__bf16*)(ws + 0);        //  8 MB  [B*S][D] bf16
    __bf16* Wtq  = (__bf16*)(ws + 8 * MB);   //  2 MB  [D][D] N-major
    __bf16* Wtk  = (__bf16*)(ws + 10 * MB);  //  2 MB
    __bf16* Wtv  = (__bf16*)(ws + 12 * MB);  //  2 MB
    __bf16* Wto  = (__bf16*)(ws + 14 * MB);  //  2 MB
    __bf16* Qb   = (__bf16*)(ws + 16 * MB);  //  8 MB  [BH][S][64]
    __bf16* Kb   = (__bf16*)(ws + 24 * MB);  //  8 MB  [BH][S][64]
    __bf16* Vt   = (__bf16*)(ws + 32 * MB);  //  8 MB  [BH][64][S]
    __bf16* ctxb = (__bf16*)(ws + 40 * MB);  //  8 MB  [B*S][D]

    const int nx = B_ * S_ * D_;             // 4,194,304
    cvt_bf16_kernel<<<nx / 256, 256, 0, stream>>>(x, xb, nx);
    transpose_w_kernel<<<(D_ * D_) / 256, 256, 0, stream>>>(Wq, Wtq);
    transpose_w_kernel<<<(D_ * D_) / 256, 256, 0, stream>>>(Wk, Wtk);
    transpose_w_kernel<<<(D_ * D_) / 256, 256, 0, stream>>>(Wv, Wtv);
    transpose_w_kernel<<<(D_ * D_) / 256, 256, 0, stream>>>(Wo, Wto);

    // 3 mats * 64 row tiles * 16 col tiles = 3072 waves / 8 per block
    qkv_gemm_kernel<<<384, 256, 0, stream>>>(xb, Wtq, Wtk, Wtv, bq, bk, bv,
                                             Qb, Kb, Vt);
    // B*H*(S/16) = 4096 waves / 8 per block
    attn_kernel<<<512, 256, 0, stream>>>(Qb, Kb, Vt, ctxb);
    // 64 * 16 = 1024 waves / 8 per block
    oproj_kernel<<<128, 256, 0, stream>>>(ctxb, Wto, bo, out);
}
